// LAVAMemory_21723944583235
// MI455X (gfx1250) — matrix operand order repaired and checked by reading, not compiled, not run on hardware
//
#include <hip/hip_runtime.h>
#include <cstdint>
#include <cstddef>

// ---------------------------------------------------------------------------
// Problem constants (reference: B=4, S=4096, H=2048, L=128, K=4)
// ---------------------------------------------------------------------------
#define H_DIM   2048
#define L_SLOTS 128
#define TOPK    4

#define TILE_M 128
#define TILE_N 128
#define TILE_K 32
#define LDSS   40   // padded LDS row stride in bf16 elems (80 B): 64B data + 16B pad

typedef __attribute__((ext_vector_type(16))) __bf16 v16bf;
typedef __attribute__((ext_vector_type(8)))  float  v8f;
typedef __attribute__((ext_vector_type(4)))  __bf16 bf16x4;
typedef __attribute__((ext_vector_type(4)))  unsigned int u32x4;
typedef __attribute__((ext_vector_type(8)))  int  i32x8;
typedef __attribute__((ext_vector_type(4)))  int  i32x4;

union V8FU { v8f v; float f[8]; };

// flat->LDS offset (LDS aperture: low 32 bits are the LDS byte address)
static __device__ __forceinline__ unsigned lds_off(const void* p) {
  return (unsigned)(uintptr_t)p;
}

// ---------------------------------------------------------------------------
// TDM: DMA one [tile_dim1=128 rows x TILE_K=32 elems] bf16 tile into LDS.
// D# per CDNA5 ISA ch.8: 2D tensor (groups 2/3 zero), data_size=2B,
// pad_enable: insert 4 DWORDs (16B) after every 16 DWORDs (64B row)
// -> LDS row stride 80B == LDSS bf16 elems.
// Toolchain uses the 6-arg builtin: (g0, g1, g2, g3, g4, cpol).
// ---------------------------------------------------------------------------
static __device__ __forceinline__ void tdm_load_tile_bf16(
    unsigned lds_addr, const __bf16* gtile, int tdim0, int tdim1, int stride0) {
  const unsigned long long ga = (unsigned long long)(uintptr_t)gtile;
  u32x4 g0;
  g0[0] = 1u;                                   // count=1, user descriptor
  g0[1] = lds_addr;                             // lds_addr [63:32]
  g0[2] = (unsigned)(ga & 0xFFFFFFFFu);         // global_addr[31:0]
  g0[3] = (unsigned)((ga >> 32) & 0x01FFFFFFu)  // global_addr[56:32]
          | (2u << 30);                         // type=2 ("image")
  i32x8 g1;
  g1[0] = (int)((1u << 16)    // data_size = 1 -> 2 bytes
              | (1u << 20)    // pad_enable
              | (3u << 22)    // pad_interval: 2<<3 = 16 DWORDs (64B)
              | (3u << 25));  // pad_amount: 3+1 = 4 DWORDs (16B)
  g1[1] = (int)(((unsigned)tdim0 & 0xFFFFu) << 16);             // tensor_dim0 lo
  g1[2] = (int)((((unsigned)tdim0 >> 16) & 0xFFFFu)
              | (((unsigned)tdim1 & 0xFFFFu) << 16));           // dim0 hi | dim1 lo
  g1[3] = (int)((((unsigned)tdim1 >> 16) & 0xFFFFu)
              | ((unsigned)TILE_K << 16));                      // dim1 hi | tile_dim0
  g1[4] = (int)(TILE_M);                                        // tile_dim1=128, tile_dim2=0
  g1[5] = stride0;                                              // tensor_dim0_stride lo
  g1[6] = 0;                                                    // stride0 hi | stride1 lo
  g1[7] = 0;
  const i32x4 z4 = {0, 0, 0, 0};
  const i32x8 z8 = {0, 0, 0, 0, 0, 0, 0, 0};
  __builtin_amdgcn_tensor_load_to_lds(g0, g1, z4, z4, z8, 0);
}

// ---------------------------------------------------------------------------
// fp32 -> bf16 bulk conversion (vectorized float4 -> bf16x4)
// ---------------------------------------------------------------------------
__global__ __launch_bounds__(256) void cvt_f32_bf16(const float* __restrict__ in,
                                                    __bf16* __restrict__ out,
                                                    int n4) {
  int i = blockIdx.x * 256 + threadIdx.x;
  if (i >= n4) return;
  float4 v = reinterpret_cast<const float4*>(in)[i];
  bf16x4 o;
  o[0] = (__bf16)v.x; o[1] = (__bf16)v.y; o[2] = (__bf16)v.z; o[3] = (__bf16)v.w;
  reinterpret_cast<bf16x4*>(out)[i] = o;
}

// ---------------------------------------------------------------------------
// addr_norm = addresses / max(||row||, 1e-8), emitted as bf16
// ---------------------------------------------------------------------------
__global__ __launch_bounds__(256) void norm_addr_rows(const float* __restrict__ addrs,
                                                      __bf16* __restrict__ anb) {
  __shared__ float red[256];
  const int r = blockIdx.x, t = threadIdx.x;
  const float* row = addrs + (size_t)r * H_DIM;
  float s = 0.f;
  for (int i = t; i < H_DIM; i += 256) { float v = row[i]; s += v * v; }
  red[t] = s;
  __syncthreads();
  for (int o = 128; o > 0; o >>= 1) {
    if (t < o) red[t] += red[t + o];
    __syncthreads();
  }
  const float rinv = 1.f / fmaxf(sqrtf(red[0]), 1e-8f);
  for (int i = t; i < H_DIM; i += 256)
    anb[(size_t)r * H_DIM + i] = (__bf16)(row[i] * rinv);
}

// ---------------------------------------------------------------------------
// C[M,N] = A[M,K] * W[N,K]^T  (bf16 WMMA, f32 accum), TDM-fed double buffer.
// Block tile 128x128, 8 waves, wave tile 32x64 (2x4 of 16x16).
// Wave 0 drives the TDM; one barrier per K-step.
// ---------------------------------------------------------------------------
template <bool OUT_BF16>
__global__ __launch_bounds__(256) void gemm_abt(const __bf16* __restrict__ A,
                                                const __bf16* __restrict__ W,
                                                void* __restrict__ Cout,
                                                int M, int N, int Kd) {
  __shared__ __bf16 As[2][TILE_M * LDSS];
  __shared__ __bf16 Bs[2][TILE_N * LDSS];

  const int t    = threadIdx.x;
  const int wave = t >> 5;
  const int lane = t & 31;
  const int hl   = lane >> 4;
  const int ln   = lane & 15;
  const int m0   = blockIdx.y * TILE_M;
  const int n0   = blockIdx.x * TILE_N;
  const int wm   = (wave & 3) * 32;   // 4 waves along M
  const int wn   = (wave >> 2) * 64;  // 2 waves along N

  const unsigned ldsA[2] = { lds_off(&As[0][0]), lds_off(&As[1][0]) };
  const unsigned ldsB[2] = { lds_off(&Bs[0][0]), lds_off(&Bs[1][0]) };
  const __bf16* Abase = A + (size_t)m0 * Kd;
  const __bf16* Bbase = W + (size_t)n0 * Kd;

  const v8f vzero = {0.f, 0.f, 0.f, 0.f, 0.f, 0.f, 0.f, 0.f};
  v8f acc[2][4];
#pragma unroll
  for (int i = 0; i < 2; ++i)
#pragma unroll
    for (int j = 0; j < 4; ++j) acc[i][j] = vzero;

  // prologue: DMA K-slab 0 into buffer 0
  if (wave == 0) {
    tdm_load_tile_bf16(ldsA[0], Abase, Kd, M, Kd);
    tdm_load_tile_bf16(ldsB[0], Bbase, Kd, N, Kd);
  }

  const int Ks = Kd / TILE_K;
  int buf = 0;
  for (int kt = 0; kt < Ks; ++kt) {
    if (wave == 0) __builtin_amdgcn_s_wait_tensorcnt((short)0);
    __syncthreads();  // all waves: tile in As/Bs[buf] is ready; prev reads drained
    if (wave == 0 && (kt + 1) < Ks) {
      const size_t koff = (size_t)(kt + 1) * TILE_K;
      tdm_load_tile_bf16(ldsA[buf ^ 1], Abase + koff, Kd, M, Kd);
      tdm_load_tile_bf16(ldsB[buf ^ 1], Bbase + koff, Kd, N, Kd);
    }

    // A frag: lane ln holds row, K = {8*hl..} and {16+8*hl..}
    v16bf af[2];
#pragma unroll
    for (int mt = 0; mt < 2; ++mt) {
      const __bf16* p = &As[buf][(wm + mt * 16 + ln) * LDSS];
      uint4* fp = reinterpret_cast<uint4*>(&af[mt]);
      fp[0] = *reinterpret_cast<const uint4*>(p + 8 * hl);
      fp[1] = *reinterpret_cast<const uint4*>(p + 16 + 8 * hl);
    }
    // B frag: lane ln holds column (W row), K = {8*hl..} and {16+8*hl..}
    v16bf bfr[4];
#pragma unroll
    for (int nt = 0; nt < 4; ++nt) {
      const __bf16* p = &Bs[buf][(wn + nt * 16 + ln) * LDSS];
      uint4* fp = reinterpret_cast<uint4*>(&bfr[nt]);
      fp[0] = *reinterpret_cast<const uint4*>(p + 8 * hl);
      fp[1] = *reinterpret_cast<const uint4*>(p + 16 + 8 * hl);
    }

#pragma unroll
    for (int mt = 0; mt < 2; ++mt)
#pragma unroll
      for (int nt = 0; nt < 4; ++nt)
        acc[mt][nt] = __builtin_amdgcn_wmma_f32_16x16x32_bf16(
            false, af[mt], false, bfr[nt], (short)0, acc[mt][nt], false, false);

    buf ^= 1;
  }

  // epilogue: C/D layout -> element (m = v + 8*hl, n = ln) per 16x16 tile
#pragma unroll
  for (int mt = 0; mt < 2; ++mt)
#pragma unroll
    for (int nt = 0; nt < 4; ++nt) {
      V8FU u; u.v = acc[mt][nt];
#pragma unroll
      for (int v = 0; v < 8; ++v) {
        const int m = m0 + wm + mt * 16 + hl * 8 + v;
        const int n = n0 + wn + nt * 16 + ln;
        if (OUT_BF16)
          reinterpret_cast<__bf16*>(Cout)[(size_t)m * N + n] = (__bf16)u.f[v];
        else
          reinterpret_cast<float*>(Cout)[(size_t)m * N + n] = u.f[v];
      }
    }
}

// ---------------------------------------------------------------------------
// scores[M,L] = (q . addr_norm^T) / max(||q row||, 1e-6)
// TDM-fed; N = L = 128 (full), wave tile 16x128 (1x8),
// fused per-row sum-of-squares reduction from the staged A tile.
// ---------------------------------------------------------------------------
__global__ __launch_bounds__(256) void score_kernel(const __bf16* __restrict__ Q,
                                                    const __bf16* __restrict__ AN,
                                                    float* __restrict__ scores,
                                                    int Mtot, int Kd) {
  __shared__ __bf16 As[2][TILE_M * LDSS];
  __shared__ __bf16 Bs[2][L_SLOTS * LDSS];
  __shared__ float red[256];
  __shared__ float rinvS[TILE_M];

  const int t    = threadIdx.x;
  const int wave = t >> 5;
  const int lane = t & 31;
  const int hl   = lane >> 4;
  const int ln   = lane & 15;
  const int m0   = blockIdx.x * TILE_M;
  const int srow = t & 127;        // sumsq worker: row in tile
  const int scol = (t >> 7) * 16;  // sumsq worker: half of the 32-wide slab

  const unsigned ldsA[2] = { lds_off(&As[0][0]), lds_off(&As[1][0]) };
  const unsigned ldsB[2] = { lds_off(&Bs[0][0]), lds_off(&Bs[1][0]) };
  const __bf16* Abase = Q + (size_t)m0 * Kd;
  const __bf16* Bbase = AN;

  const v8f vzero = {0.f, 0.f, 0.f, 0.f, 0.f, 0.f, 0.f, 0.f};
  v8f acc[8];
#pragma unroll
  for (int j = 0; j < 8; ++j) acc[j] = vzero;
  float ssq = 0.f;

  if (wave == 0) {
    tdm_load_tile_bf16(ldsA[0], Abase, Kd, Mtot, Kd);
    tdm_load_tile_bf16(ldsB[0], Bbase, Kd, L_SLOTS, Kd);
  }

  const int Ks = Kd / TILE_K;
  int buf = 0;
  for (int kt = 0; kt < Ks; ++kt) {
    if (wave == 0) __builtin_amdgcn_s_wait_tensorcnt((short)0);
    __syncthreads();
    if (wave == 0 && (kt + 1) < Ks) {
      const size_t koff = (size_t)(kt + 1) * TILE_K;
      tdm_load_tile_bf16(ldsA[buf ^ 1], Abase + koff, Kd, Mtot, Kd);
      tdm_load_tile_bf16(ldsB[buf ^ 1], Bbase + koff, Kd, L_SLOTS, Kd);
    }

    // fused ||q||^2 partials from the staged A tile (each element once)
#pragma unroll
    for (int i = 0; i < 16; ++i) {
      float v = (float)As[buf][srow * LDSS + scol + i];
      ssq += v * v;
    }

    v16bf afr;
    {
      const __bf16* p = &As[buf][(wave * 16 + ln) * LDSS];
      uint4* fp = reinterpret_cast<uint4*>(&afr);
      fp[0] = *reinterpret_cast<const uint4*>(p + 8 * hl);
      fp[1] = *reinterpret_cast<const uint4*>(p + 16 + 8 * hl);
    }
    v16bf bfr[8];
#pragma unroll
    for (int nt = 0; nt < 8; ++nt) {
      const __bf16* p = &Bs[buf][(nt * 16 + ln) * LDSS];
      uint4* fp = reinterpret_cast<uint4*>(&bfr[nt]);
      fp[0] = *reinterpret_cast<const uint4*>(p + 8 * hl);
      fp[1] = *reinterpret_cast<const uint4*>(p + 16 + 8 * hl);
    }
#pragma unroll
    for (int nt = 0; nt < 8; ++nt)
      acc[nt] = __builtin_amdgcn_wmma_f32_16x16x32_bf16(
          false, afr, false, bfr[nt], (short)0, acc[nt], false, false);

    buf ^= 1;
  }

  // combine the two half-row partials, build 1/max(||q||,1e-6)
  red[t] = ssq;
  __syncthreads();
  if (t < 128) {
    float tot = red[t] + red[t + 128];
    rinvS[t] = 1.f / fmaxf(sqrtf(tot), 1e-6f);
  }
  __syncthreads();

#pragma unroll
  for (int nt = 0; nt < 8; ++nt) {
    V8FU u; u.v = acc[nt];
#pragma unroll
    for (int v = 0; v < 8; ++v) {
      const int ml = wave * 16 + hl * 8 + v;
      const int n  = nt * 16 + ln;
      scores[(size_t)(m0 + ml) * L_SLOTS + n] = u.f[v] * rinvS[ml];
    }
  }
}

// ---------------------------------------------------------------------------
// Per-row top-4 + softmax + weighted gather of contents -> mem (bf16)
// 8 rows per 256-thread block; contents is 1 MB -> L2 resident.
// ---------------------------------------------------------------------------
__global__ __launch_bounds__(256) void topk_gather(const float* __restrict__ scores,
                                                   const float* __restrict__ contents,
                                                   __bf16* __restrict__ memb) {
  __shared__ int   sidx[8][TOPK];
  __shared__ float satt[8][TOPK];
  const int t  = threadIdx.x;
  const int rb = blockIdx.x * 8;

  if (t < 8) {
    const float* s = scores + (size_t)(rb + t) * L_SLOTS;
    float v0 = -1e30f, v1 = -1e30f, v2 = -1e30f, v3 = -1e30f;
    int   i0 = 0, i1 = 0, i2 = 0, i3 = 0;
    for (int l = 0; l < L_SLOTS; ++l) {
      float x = s[l];
      if (x > v0)      { v3=v2;i3=i2; v2=v1;i2=i1; v1=v0;i1=i0; v0=x;i0=l; }
      else if (x > v1) { v3=v2;i3=i2; v2=v1;i2=i1; v1=x;i1=l; }
      else if (x > v2) { v3=v2;i3=i2; v2=x;i2=l; }
      else if (x > v3) { v3=x;i3=l; }
    }
    float e0 = __expf(v0 - v0), e1 = __expf(v1 - v0);
    float e2 = __expf(v2 - v0), e3 = __expf(v3 - v0);
    float inv = 1.f / (e0 + e1 + e2 + e3);
    satt[t][0] = e0 * inv; satt[t][1] = e1 * inv;
    satt[t][2] = e2 * inv; satt[t][3] = e3 * inv;
    sidx[t][0] = i0; sidx[t][1] = i1; sidx[t][2] = i2; sidx[t][3] = i3;
  }
  __syncthreads();

  const int r = t >> 5, lane = t & 31;
  const float4* c0 = reinterpret_cast<const float4*>(contents + (size_t)sidx[r][0] * H_DIM);
  const float4* c1 = reinterpret_cast<const float4*>(contents + (size_t)sidx[r][1] * H_DIM);
  const float4* c2 = reinterpret_cast<const float4*>(contents + (size_t)sidx[r][2] * H_DIM);
  const float4* c3 = reinterpret_cast<const float4*>(contents + (size_t)sidx[r][3] * H_DIM);
  const float a0 = satt[r][0], a1 = satt[r][1], a2 = satt[r][2], a3 = satt[r][3];
  __bf16* outrow = memb + (size_t)(rb + r) * H_DIM;

#pragma unroll 4
  for (int j = 0; j < 16; ++j) {
    int c = lane + j * 32;  // float4 index, 512 per row
    float4 x0 = c0[c], x1 = c1[c], x2 = c2[c], x3 = c3[c];
    float rx = a0 * x0.x + a1 * x1.x + a2 * x2.x + a3 * x3.x;
    float ry = a0 * x0.y + a1 * x1.y + a2 * x2.y + a3 * x3.y;
    float rz = a0 * x0.z + a1 * x1.z + a2 * x2.z + a3 * x3.z;
    float rw = a0 * x0.w + a1 * x1.w + a2 * x2.w + a3 * x3.w;
    bf16x4 o;
    o[0] = (__bf16)rx; o[1] = (__bf16)ry; o[2] = (__bf16)rz; o[3] = (__bf16)rw;
    reinterpret_cast<bf16x4*>(outrow)[c] = o;
  }
}

// ---------------------------------------------------------------------------
// Host-side launcher
// ---------------------------------------------------------------------------
extern "C" void kernel_launch(void* const* d_in, const int* in_sizes, int n_in,
                              void* d_out, int out_size, void* d_ws, size_t ws_size,
                              hipStream_t stream) {
  (void)n_in; (void)out_size; (void)ws_size;
  const float* x         = (const float*)d_in[0];
  const float* W_addr    = (const float*)d_in[1];
  const float* W_read    = (const float*)d_in[2];
  const float* addresses = (const float*)d_in[3];
  const float* contents  = (const float*)d_in[4];
  // d_in[5] = top_k (==4 == TOPK; min(4,128)=4) — compiled in.

  const int M = in_sizes[0] / H_DIM;  // B*S = 16384

  // workspace layout (bytes)
  char* ws = (char*)d_ws;
  const size_t szMH = (size_t)M * H_DIM * sizeof(__bf16);       // 67 MB
  const size_t szHH = (size_t)H_DIM * H_DIM * sizeof(__bf16);   // 8.4 MB
  const size_t szLH = (size_t)L_SLOTS * H_DIM * sizeof(__bf16); // 0.5 MB
  __bf16* xb     = (__bf16*)(ws);
  __bf16* qb     = (__bf16*)(ws + szMH);
  __bf16* wab    = (__bf16*)(ws + 2 * szMH);
  __bf16* wrb    = (__bf16*)(ws + 2 * szMH + szHH);
  __bf16* anb    = (__bf16*)(ws + 2 * szMH + 2 * szHH);
  float*  scores = (float*) (ws + 2 * szMH + 2 * szHH + szLH);
  __bf16* memb   = xb;  // x is dead after GEMM1 -> reuse for mem

  // stage 0: precision conversion + address cache normalization
  {
    int n4 = in_sizes[0] / 4;
    cvt_f32_bf16<<<dim3((n4 + 255) / 256), 256, 0, stream>>>(x, xb, n4);
  }
  {
    int n4 = in_sizes[1] / 4;
    cvt_f32_bf16<<<dim3((n4 + 255) / 256), 256, 0, stream>>>(W_addr, wab, n4);
  }
  {
    int n4 = in_sizes[2] / 4;
    cvt_f32_bf16<<<dim3((n4 + 255) / 256), 256, 0, stream>>>(W_read, wrb, n4);
  }
  norm_addr_rows<<<dim3(L_SLOTS), 256, 0, stream>>>(addresses, anb);

  // stage 1: q = x @ W_addr^T   (bf16 WMMA, bf16 out, TDM-fed)
  gemm_abt<true><<<dim3(H_DIM / TILE_N, M / TILE_M), 256, 0, stream>>>(
      xb, wab, qb, M, H_DIM, H_DIM);

  // stage 2: scores = normalize(q) . addr_norm^T  (fused row-norm, TDM-fed)
  score_kernel<<<dim3(M / TILE_M), 256, 0, stream>>>(qb, anb, scores, M, H_DIM);

  // stage 3: top-4 + softmax + gather -> mem (bf16, reuses x buffer)
  topk_gather<<<dim3(M / 8), 256, 0, stream>>>(scores, contents, memb);

  // stage 4: out = mem @ W_read^T  (bf16 WMMA, fp32 out, TDM-fed)
  gemm_abt<false><<<dim3(H_DIM / TILE_N, M / TILE_M), 256, 0, stream>>>(
      memb, wrb, d_out, M, H_DIM, H_DIM);
}
